// MetalWallQEQ_69466801045970
// MI455X (gfx1250) — matrix operand order repaired and checked by reading, not compiled, not run on hardware
//
#include <hip/hip_runtime.h>
#include <hip/hip_bf16.h>

// ---------------------------------------------------------------------------
// MetalWall QEQ on MI455X (gfx1250, wave32).
//   Cw/Sw = sqrt(kfac) * cos/sin(r_metal . k)  stored f16  [2048 x 15648]
//   A     = pref * (Cw Cw^T + Sw Sw^T)         f32, via v_wmma_f32_16x16x32_f16
//           (128x64 workgroup tile, 4x2 waves, 32x32 per wave = 2x2 WMMA tiles)
//   KKT solved with Schur complement + 2-column CG (A is SPD with +J diag).
// ---------------------------------------------------------------------------

typedef __attribute__((ext_vector_type(16))) _Float16 v16h;
typedef __attribute__((ext_vector_type(8)))  _Float16 v8h;
typedef __attribute__((ext_vector_type(8)))  float    v8f;

#define N_ATOMS   4096
#define NM        2048        // metal atoms = first 2048 rows
#define NKDIM     12          // ceil(24/2)
#define KAX       25          // 2*12+1
#define KTOT      15624       // 25^3 - 1
#define KZERO     7812        // linear index of (0,0,0) in full 25^3 grid
#define KP        15648       // KTOT padded to multiple of 32 (489*32)
#define KSTEP     32
#define LDA_P     40          // LDS pitch (halves) for A tiles (128x32); 80B rows
#define LDB_P     72          // LDS pitch (halves) for transposed B tiles (32x64); 144B rows
#define CG_ITERS  256

// ---------------- setup: reciprocal lattice, volume, J ----------------------
__global__ void k_setup(const float* __restrict__ cell,
                        const float* __restrict__ Jraw,
                        float* __restrict__ hdr) {
    float a00=cell[0],a01=cell[1],a02=cell[2];
    float a10=cell[3],a11=cell[4],a12=cell[5];
    float a20=cell[6],a21=cell[7],a22=cell[8];
    float det = a00*(a11*a22-a12*a21) - a01*(a10*a22-a12*a20) + a02*(a10*a21-a11*a20);
    float id = 1.0f/det;
    float inv[3][3];
    inv[0][0]=(a11*a22-a12*a21)*id; inv[0][1]=(a02*a21-a01*a22)*id; inv[0][2]=(a01*a12-a02*a11)*id;
    inv[1][0]=(a12*a20-a10*a22)*id; inv[1][1]=(a00*a22-a02*a20)*id; inv[1][2]=(a02*a10-a00*a12)*id;
    inv[2][0]=(a10*a21-a11*a20)*id; inv[2][1]=(a01*a20-a00*a21)*id; inv[2][2]=(a00*a11-a01*a10)*id;
    const float twopi = 6.2831853071795864769f;
    for (int a = 0; a < 3; ++a)
        for (int d = 0; d < 3; ++d)
            hdr[a*3+d] = twopi * inv[d][a];      // rows of 2*pi*inv(cell)^T
    float vol = fabsf(det);
    hdr[9]  = 2.0f*twopi / vol;      // pref = 4*pi/V
    hdr[10] = Jraw[0]*Jraw[0];       // hardness J
}

__device__ inline void k_vec_of(int k, const float* hdr, float kv[3], bool& valid) {
    valid = (k < KTOT);
    int lin = k + (k >= KZERO ? 1 : 0);
    float gz = (float)(lin % KAX - NKDIM);
    float gy = (float)((lin / KAX) % KAX - NKDIM);
    float gx = (float)(lin / (KAX*KAX) - NKDIM);
    kv[0] = gx*hdr[0] + gy*hdr[3] + gz*hdr[6];
    kv[1] = gx*hdr[1] + gy*hdr[4] + gz*hdr[7];
    kv[2] = gx*hdr[2] + gy*hdr[5] + gz*hdr[8];
}

// ---------------- Cw/Sw = sqrt(kfac) * cos/sin(r_metal . k) -----------------
__global__ __launch_bounds__(256)
void k_build_cs(const float* __restrict__ pos, const float* __restrict__ hdr,
                _Float16* __restrict__ Cw, _Float16* __restrict__ Sw) {
    size_t idx = (size_t)blockIdx.x * blockDim.x + threadIdx.x;   // NM*KP % 256 == 0
    int i = (int)(idx / KP);
    int k = (int)(idx % KP);
    float kv[3]; bool valid;
    k_vec_of(k, hdr, kv, valid);
    _Float16 c = (_Float16)0.f, s = (_Float16)0.f;
    if (valid) {
        float k2 = kv[0]*kv[0] + kv[1]*kv[1] + kv[2]*kv[2];
        float w  = sqrtf(__expf(-0.5f * k2) / k2);   // sqrt(kfac), sigma = 1
        float ph = pos[3*i]*kv[0] + pos[3*i+1]*kv[1] + pos[3*i+2]*kv[2];
        float sn, cs; __sincosf(ph, &sn, &cs);
        c = (_Float16)(w * cs);
        s = (_Float16)(w * sn);
    }
    Cw[idx] = c; Sw[idx] = s;
}

// ---------------- electrolyte structure factor (weighted) -------------------
__global__ __launch_bounds__(256)
void k_struct_fac(const float* __restrict__ pos, const float* __restrict__ q,
                  const float* __restrict__ hdr,
                  float* __restrict__ Srw, float* __restrict__ Siw) {
    int k = blockIdx.x * blockDim.x + threadIdx.x;
    if (k >= KP) return;
    float kv[3]; bool valid;
    k_vec_of(k, hdr, kv, valid);
    float sr = 0.f, si = 0.f, w = 0.f;
    if (valid) {
        float k2 = kv[0]*kv[0] + kv[1]*kv[1] + kv[2]*kv[2];
        w = sqrtf(__expf(-0.5f * k2) / k2);
        for (int j = NM; j < N_ATOMS; ++j) {    // metal charges zeroed in reference
            float ph = pos[3*j]*kv[0] + pos[3*j+1]*kv[1] + pos[3*j+2]*kv[2];
            float sn, cs; __sincosf(ph, &sn, &cs);
            sr += q[j]*cs; si += q[j]*sn;
        }
    }
    Srw[k] = w * sr;    // so that  c*kfac*Sr == Cw * Srw
    Siw[k] = w * si;
}

// ---------------- A = pref*(Cw Cw^T + Sw Sw^T) via WMMA ---------------------
// Workgroup tile: 128 (rows i) x 64 (cols j); 8 waves as 4x2, each 32x32 =
// 2x2 WMMA tiles. C- and S-Gram terms accumulate into the same 4 accumulators.
__global__ __launch_bounds__(256)
void k_syrk_wmma(const _Float16* __restrict__ Cw, const _Float16* __restrict__ Sw,
                 const float* __restrict__ hdr, float* __restrict__ A) {
    __shared__ _Float16 ldsAc[128 * LDA_P];   // 128x32 row-major (+pad)
    __shared__ _Float16 ldsAs[128 * LDA_P];
    __shared__ _Float16 ldsBc[32 * LDB_P];    // transposed: [k][j], 32x64 (+pad)
    __shared__ _Float16 ldsBs[32 * LDB_P];

    const float pref = hdr[9];
    const int tid  = threadIdx.x;
    const int i0   = blockIdx.x * 128;
    const int j0   = blockIdx.y * 64;
    const int wave = tid >> 5, lane = tid & 31;
    const int iw   = wave >> 1, jw = wave & 1;     // 4x2 waves

    // A-operand fragment source (ISA 16-bit A 16x32 layout)
    const int fm = lane & 15, fkb = (lane >> 4) * 8;
    // staging maps: A = 512 chunks/matrix -> 2 per thread; B = 256 -> 1 per thread
    const int ar0 = tid >> 2, ar1 = 64 + (tid >> 2), akc = (tid & 3) * 8;
    const int bjr = tid >> 2, bkc = (tid & 3) * 8;

    v8f acc00 = {}, acc01 = {}, acc10 = {}, acc11 = {};

    for (int k0 = 0; k0 < KP; k0 += KSTEP) {
        __syncthreads();
        // ---- stage A panels (coalesced b128 loads) ----
        {
            size_t g0 = (size_t)(i0 + ar0) * KP + k0 + akc;
            size_t g1 = (size_t)(i0 + ar1) * KP + k0 + akc;
            *(v8h*)&ldsAc[ar0 * LDA_P + akc] = *(const v8h*)&Cw[g0];
            *(v8h*)&ldsAc[ar1 * LDA_P + akc] = *(const v8h*)&Cw[g1];
            *(v8h*)&ldsAs[ar0 * LDA_P + akc] = *(const v8h*)&Sw[g0];
            *(v8h*)&ldsAs[ar1 * LDA_P + akc] = *(const v8h*)&Sw[g1];
        }
        // ---- stage B panels transposed into [k][j] ----
        {
            size_t g = (size_t)(j0 + bjr) * KP + k0 + bkc;
            v8h vc = *(const v8h*)&Cw[g];
            v8h vs = *(const v8h*)&Sw[g];
            #pragma unroll
            for (int e = 0; e < 8; ++e) {
                ldsBc[(bkc + e) * LDB_P + bjr] = vc[e];
                ldsBs[(bkc + e) * LDB_P + bjr] = vs[e];
            }
        }
        __syncthreads();

        // ---- fragments + 8 WMMA (no divergence: EXEC all ones) ----
        v16h a0, a1, b0, b1;
        v8h lo, hi;
        const int arow0 = (iw * 32 + fm) * LDA_P;
        const int arow1 = (iw * 32 + 16 + fm) * LDA_P;
        const int brow  = jw * 32 + fm;

        // ---------------- cos Gram term ----------------
        lo = *(const v8h*)&ldsAc[arow0 + fkb];
        hi = *(const v8h*)&ldsAc[arow0 + fkb + 16];
        #pragma unroll
        for (int e = 0; e < 8; ++e) { a0[e] = lo[e]; a0[e+8] = hi[e]; }
        lo = *(const v8h*)&ldsAc[arow1 + fkb];
        hi = *(const v8h*)&ldsAc[arow1 + fkb + 16];
        #pragma unroll
        for (int e = 0; e < 8; ++e) { a1[e] = lo[e]; a1[e+8] = hi[e]; }
        #pragma unroll
        for (int e = 0; e < 8; ++e) { b0[e] = ldsBc[(fkb + e) * LDB_P + brow]; b0[e+8] = ldsBc[(fkb + 16 + e) * LDB_P + brow]; }
        #pragma unroll
        for (int e = 0; e < 8; ++e) { b1[e] = ldsBc[(fkb + e) * LDB_P + brow + 16]; b1[e+8] = ldsBc[(fkb + 16 + e) * LDB_P + brow + 16]; }
        acc00 = __builtin_amdgcn_wmma_f32_16x16x32_f16(false, a0, false, b0, (short)0, acc00, false, false);
        acc01 = __builtin_amdgcn_wmma_f32_16x16x32_f16(false, a0, false, b1, (short)0, acc01, false, false);
        acc10 = __builtin_amdgcn_wmma_f32_16x16x32_f16(false, a1, false, b0, (short)0, acc10, false, false);
        acc11 = __builtin_amdgcn_wmma_f32_16x16x32_f16(false, a1, false, b1, (short)0, acc11, false, false);

        // ---------------- sin Gram term ----------------
        lo = *(const v8h*)&ldsAs[arow0 + fkb];
        hi = *(const v8h*)&ldsAs[arow0 + fkb + 16];
        #pragma unroll
        for (int e = 0; e < 8; ++e) { a0[e] = lo[e]; a0[e+8] = hi[e]; }
        lo = *(const v8h*)&ldsAs[arow1 + fkb];
        hi = *(const v8h*)&ldsAs[arow1 + fkb + 16];
        #pragma unroll
        for (int e = 0; e < 8; ++e) { a1[e] = lo[e]; a1[e+8] = hi[e]; }
        #pragma unroll
        for (int e = 0; e < 8; ++e) { b0[e] = ldsBs[(fkb + e) * LDB_P + brow]; b0[e+8] = ldsBs[(fkb + 16 + e) * LDB_P + brow]; }
        #pragma unroll
        for (int e = 0; e < 8; ++e) { b1[e] = ldsBs[(fkb + e) * LDB_P + brow + 16]; b1[e+8] = ldsBs[(fkb + 16 + e) * LDB_P + brow + 16]; }
        acc00 = __builtin_amdgcn_wmma_f32_16x16x32_f16(false, a0, false, b0, (short)0, acc00, false, false);
        acc01 = __builtin_amdgcn_wmma_f32_16x16x32_f16(false, a0, false, b1, (short)0, acc01, false, false);
        acc10 = __builtin_amdgcn_wmma_f32_16x16x32_f16(false, a1, false, b0, (short)0, acc10, false, false);
        acc11 = __builtin_amdgcn_wmma_f32_16x16x32_f16(false, a1, false, b1, (short)0, acc11, false, false);
    }

    // C/D layout: lane l, VGPR v -> M = v + 8*(l>>4), N = l&15
    const int rb = i0 + iw * 32 + ((lane >> 4) * 8);
    const int cb = j0 + jw * 32 + (lane & 15);
    #pragma unroll
    for (int v = 0; v < 8; ++v) {
        A[(size_t)(rb + v)      * NM + cb]      = pref * acc00[v];
        A[(size_t)(rb + v)      * NM + cb + 16] = pref * acc01[v];
        A[(size_t)(rb + v + 16) * NM + cb]      = pref * acc10[v];
        A[(size_t)(rb + v + 16) * NM + cb + 16] = pref * acc11[v];
    }
}

// ---------------- B = -pref*(Cw.Srw + Sw.Siw); init CG state ----------------
__global__ __launch_bounds__(256)
void k_rhs_init(const _Float16* __restrict__ Cw, const _Float16* __restrict__ Sw,
                const float* __restrict__ Srw, const float* __restrict__ Siw,
                const float* __restrict__ hdr,
                float* __restrict__ x0, float* __restrict__ x1,
                float* __restrict__ r0, float* __restrict__ r1,
                float* __restrict__ p0, float* __restrict__ p1) {
    const int lane = threadIdx.x & 31;
    const int row  = blockIdx.x * 8 + (threadIdx.x >> 5);
    size_t base = (size_t)row * KP;
    float s = 0.f;
    for (int k = lane; k < KP; k += 32)
        s += (float)Cw[base + k] * Srw[k] + (float)Sw[base + k] * Siw[k];
    #pragma unroll
    for (int off = 16; off; off >>= 1) s += __shfl_xor(s, off, 32);
    if (lane == 0) {
        float Bv = -hdr[9] * s;                 // scaling = 1
        x0[row] = 0.f; r0[row] = Bv;  p0[row] = Bv;
        x1[row] = 0.f; r1[row] = 1.f; p1[row] = 1.f;
    }
}

// ---------------- CG kernels -----------------------------------------------
__global__ __launch_bounds__(256)
void k_matvec(const float* __restrict__ A, const float* __restrict__ hdr,
              const float* __restrict__ p0, const float* __restrict__ p1,
              float* __restrict__ z0, float* __restrict__ z1) {
    const int lane = threadIdx.x & 31;
    const int row  = blockIdx.x * 8 + (threadIdx.x >> 5);
    const float* Ar = A + (size_t)row * NM;
    float s0 = 0.f, s1 = 0.f;
    for (int c = lane; c < NM; c += 32) {
        float a = Ar[c];
        s0 += a * p0[c]; s1 += a * p1[c];
    }
    #pragma unroll
    for (int off = 16; off; off >>= 1) {
        s0 += __shfl_xor(s0, off, 32);
        s1 += __shfl_xor(s1, off, 32);
    }
    if (lane == 0) {
        float J = hdr[10];
        z0[row] = s0 + J * p0[row];
        z1[row] = s1 + J * p1[row];
    }
}

__device__ inline void block_red2(float s0, float s1, float* out0, float* out1) {
    __shared__ float sm0[256], sm1[256];
    int t = threadIdx.x;
    sm0[t] = s0; sm1[t] = s1; __syncthreads();
    for (int w = 128; w; w >>= 1) {
        if (t < w) { sm0[t] += sm0[t+w]; sm1[t] += sm1[t+w]; }
        __syncthreads();
    }
    if (t == 0) { *out0 = sm0[0]; *out1 = sm1[0]; }
}

// sc: [0]=rr0 [1]=rr1 [2]=a0 [3]=a1 [4]=b0 [5]=b1 [6]=lambda
__global__ void k_rr_init(const float* r0, const float* r1, float* sc) {
    float s0 = 0.f, s1 = 0.f;
    for (int i = threadIdx.x; i < NM; i += 256) { s0 += r0[i]*r0[i]; s1 += r1[i]*r1[i]; }
    __shared__ float o0, o1;
    block_red2(s0, s1, &o0, &o1);
    if (threadIdx.x == 0) { sc[0] = o0; sc[1] = o1; }
}

__global__ void k_alpha(const float* p0, const float* p1,
                        const float* z0, const float* z1, float* sc) {
    float s0 = 0.f, s1 = 0.f;
    for (int i = threadIdx.x; i < NM; i += 256) { s0 += p0[i]*z0[i]; s1 += p1[i]*z1[i]; }
    __shared__ float o0, o1;
    block_red2(s0, s1, &o0, &o1);
    if (threadIdx.x == 0) {
        sc[2] = (o0 != 0.f) ? sc[0] / o0 : 0.f;
        sc[3] = (o1 != 0.f) ? sc[1] / o1 : 0.f;
    }
}

__global__ void k_update_xr(float* x0, float* x1, float* r0, float* r1,
                            const float* p0, const float* p1,
                            const float* z0, const float* z1, const float* sc) {
    int i = blockIdx.x * blockDim.x + threadIdx.x;
    float a0 = sc[2], a1 = sc[3];
    x0[i] += a0 * p0[i];  r0[i] -= a0 * z0[i];
    x1[i] += a1 * p1[i];  r1[i] -= a1 * z1[i];
}

__global__ void k_beta(const float* r0, const float* r1, float* sc) {
    float s0 = 0.f, s1 = 0.f;
    for (int i = threadIdx.x; i < NM; i += 256) { s0 += r0[i]*r0[i]; s1 += r1[i]*r1[i]; }
    __shared__ float o0, o1;
    block_red2(s0, s1, &o0, &o1);
    if (threadIdx.x == 0) {
        sc[4] = (sc[0] > 0.f) ? o0 / sc[0] : 0.f;
        sc[5] = (sc[1] > 0.f) ? o1 / sc[1] : 0.f;
        sc[0] = o0; sc[1] = o1;
    }
}

__global__ void k_update_p(float* p0, float* p1,
                           const float* r0, const float* r1, const float* sc) {
    int i = blockIdx.x * blockDim.x + threadIdx.x;
    p0[i] = r0[i] + sc[4] * p0[i];
    p1[i] = r1[i] + sc[5] * p1[i];
}

__global__ void k_lambda(const float* x0, const float* x1, float* sc) {
    float s0 = 0.f, s1 = 0.f;
    for (int i = threadIdx.x; i < NM; i += 256) { s0 += x0[i]; s1 += x1[i]; }
    __shared__ float o0, o1;
    block_red2(s0, s1, &o0, &o1);
    if (threadIdx.x == 0) sc[6] = (o1 != 0.f) ? o0 / o1 : 0.f;
}

__global__ void k_write_out(const float* x0, const float* x1, const float* sc,
                            const float* __restrict__ q, float* __restrict__ out) {
    int i = blockIdx.x * blockDim.x + threadIdx.x;
    if (i < NM) out[i] = x0[i] - sc[6] * x1[i];   // q_sol / scaling (scaling = 1)
    else        out[i] = q[i];
}

// ---------------------------------------------------------------------------
extern "C" void kernel_launch(void* const* d_in, const int* in_sizes, int n_in,
                              void* d_out, int out_size, void* d_ws, size_t ws_size,
                              hipStream_t stream) {
    const float* pos   = (const float*)d_in[0];   // [4096,3]
    const float* cell  = (const float*)d_in[1];   // [1,3,3]
    const float* q     = (const float*)d_in[2];   // [4096,1]
    const float* Jraw  = (const float*)d_in[3];   // [1]
    (void)in_sizes; (void)n_in; (void)ws_size; (void)out_size;

    // carve workspace (256B-aligned slabs)
    char* ws = (char*)d_ws;
    size_t off = 0;
    auto carve = [&](size_t bytes) {
        char* p = ws + off;
        off += (bytes + 255) & ~(size_t)255;
        return p;
    };
    float*    hdr = (float*)   carve(16 * sizeof(float));
    _Float16* Cw  = (_Float16*)carve((size_t)NM * KP * sizeof(_Float16));
    _Float16* Sw  = (_Float16*)carve((size_t)NM * KP * sizeof(_Float16));
    float*    A   = (float*)   carve((size_t)NM * NM * sizeof(float));
    float*    Srw = (float*)   carve(KP * sizeof(float));
    float*    Siw = (float*)   carve(KP * sizeof(float));
    float*    x0  = (float*)   carve(NM * sizeof(float));
    float*    x1  = (float*)   carve(NM * sizeof(float));
    float*    r0  = (float*)   carve(NM * sizeof(float));
    float*    r1  = (float*)   carve(NM * sizeof(float));
    float*    p0  = (float*)   carve(NM * sizeof(float));
    float*    p1  = (float*)   carve(NM * sizeof(float));
    float*    z0  = (float*)   carve(NM * sizeof(float));
    float*    z1  = (float*)   carve(NM * sizeof(float));
    float*    sc  = (float*)   carve(16 * sizeof(float));
    float*    out = (float*)d_out;

    k_setup<<<1, 1, 0, stream>>>(cell, Jraw, hdr);

    k_build_cs<<<(unsigned)((size_t)NM * KP / 256), 256, 0, stream>>>(pos, hdr, Cw, Sw);

    k_struct_fac<<<(KP + 255) / 256, 256, 0, stream>>>(pos, q, hdr, Srw, Siw);

    k_syrk_wmma<<<dim3(NM / 128, NM / 64), 256, 0, stream>>>(Cw, Sw, hdr, A);

    k_rhs_init<<<NM / 8, 256, 0, stream>>>(Cw, Sw, Srw, Siw, hdr, x0, x1, r0, r1, p0, p1);

    k_rr_init<<<1, 256, 0, stream>>>(r0, r1, sc);

    for (int it = 0; it < CG_ITERS; ++it) {
        k_matvec<<<NM / 8, 256, 0, stream>>>(A, hdr, p0, p1, z0, z1);
        k_alpha<<<1, 256, 0, stream>>>(p0, p1, z0, z1, sc);
        k_update_xr<<<NM / 256, 256, 0, stream>>>(x0, x1, r0, r1, p0, p1, z0, z1, sc);
        k_beta<<<1, 256, 0, stream>>>(r0, r1, sc);
        k_update_p<<<NM / 256, 256, 0, stream>>>(p0, p1, r0, r1, sc);
    }

    k_lambda<<<1, 256, 0, stream>>>(x0, x1, sc);
    k_write_out<<<N_ATOMS / 256, 256, 0, stream>>>(x0, x1, sc, q, out);
}
